// GRU_48679159333115
// MI455X (gfx1250) — matrix operand order repaired and checked by reading, not compiled, not run on hardware
//
#include <hip/hip_runtime.h>

// ---------------------------------------------------------------------------
// Persistent 2-layer GRU for MI455X (gfx1250, wave32, WMMA bf16).
// Serial over T=512 -> latency bound: one persistent kernel, 513 rounds,
// one grid barrier per round; layer0@t and layer1@(t-1) run concurrently
// (blocks 0..15 / 16..31). Weights bf16 in L2 (~1.1 MB). Hidden state kept
// in BOTH fp32 (for the elementwise GRU update) and bf16 mirrors (direct
// WMMA A-fragment loads, no per-round conversion). Only x (K=128 of 384 in
// layer0) is converted fp32->bf16 in-register via v_perm_b32 packing.
// ---------------------------------------------------------------------------

typedef __attribute__((ext_vector_type(16))) __bf16   v16bf;
typedef __attribute__((ext_vector_type(8)))  float    v8f;
typedef __attribute__((ext_vector_type(8)))  unsigned v8u;

#define T_STEPS 512
#define BB      256
#define II      128
#define HH      256
#define NBLK    32          // 16 blocks per layer

// workspace element counts
#define N_RZ0 (2 * HH * (II + HH))   // 512*384
#define N_N0  (HH * II)              // 256*128
#define N_RZ1 (2 * HH * (HH + HH))   // 512*512
#define N_N1  (HH * HH)              // 256*256
#define N_H   (BB * HH)              // 256*256

__device__ __forceinline__ __bf16 f2bf(float f) {
  unsigned u = __builtin_bit_cast(unsigned, f);
  u = u + 0x7FFFu + ((u >> 16) & 1u);               // round-to-nearest-even
  unsigned short s = (unsigned short)(u >> 16);
  return __builtin_bit_cast(__bf16, s);
}

// pack two fp32 into two bf16 (1 add each + one v_perm_b32)
__device__ __forceinline__ unsigned pack2bf(float f0, float f1) {
  const unsigned u0 = __builtin_bit_cast(unsigned, f0) + 0x8000u;
  const unsigned u1 = __builtin_bit_cast(unsigned, f1) + 0x8000u;
  return __builtin_amdgcn_perm(u1, u0, 0x07060302u); // {u1.hi16, u0.hi16}
}

__device__ __forceinline__ float fsigmoid(float x) {
  return 1.0f / (1.0f + __expf(-x));
}
__device__ __forceinline__ float ftanh_fast(float x) {
  return 1.0f - 2.0f / (__expf(2.0f * x) + 1.0f);
}

// A fragment (16x32 bf16 MxK): lane l (m=l&15, kh=l>>4):
//   e in [0,16): K = kk + kh*16 + e
__device__ __forceinline__ v16bf frag_from_f32(const float* __restrict__ p0,
                                               const float* __restrict__ p1) {
  const float4 a = ((const float4*)p0)[0];
  const float4 b = ((const float4*)p0)[1];
  const float4 c = ((const float4*)p1)[0];
  const float4 d = ((const float4*)p1)[1];
  v8u t;
  t[0] = pack2bf(a.x, a.y); t[1] = pack2bf(a.z, a.w);
  t[2] = pack2bf(b.x, b.y); t[3] = pack2bf(b.z, b.w);
  t[4] = pack2bf(c.x, c.y); t[5] = pack2bf(c.z, c.w);
  t[6] = pack2bf(d.x, d.y); t[7] = pack2bf(d.z, d.w);
  return __builtin_bit_cast(v16bf, t);
}

__device__ __forceinline__ v16bf frag_from_bf16(const __bf16* __restrict__ p0,
                                                const __bf16* __restrict__ p1) {
  const uint4 lo = *(const uint4*)p0;   // 8 contiguous bf16, 16B aligned
  const uint4 hi = *(const uint4*)p1;
  v8u t;
  t[0] = lo.x; t[1] = lo.y; t[2] = lo.z; t[3] = lo.w;
  t[4] = hi.x; t[5] = hi.y; t[6] = hi.z; t[7] = hi.w;
  return __builtin_bit_cast(v16bf, t);
}

__device__ __forceinline__ v8f wmma_bf16(v16bf a, v16bf b, v8f c) {
  return __builtin_amdgcn_wmma_f32_16x16x32_bf16(false, a, false, b,
                                                 (short)0, c, false, false);
}

// One GRU cell for one (16-row batch tile) x (two 16-col tiles per wave).
// A = [a0 | a1]: a0 is fp32 (A0F32=true, x) or bf16 mirror; a1 always bf16.
// wrz: bf16 [2H x ktot] row-major; wn: bf16 [H x k0len] row-major.
template <bool A0F32>
__device__ __forceinline__ void gru_cell_tile(
    const void* __restrict__ a0v, int lda0, int k0len,
    const __bf16* __restrict__ a1bf, int lda1, int ktot,
    const __bf16* __restrict__ wrz, int ldrz,
    const __bf16* __restrict__ wn,  int ldn,
    const float* __restrict__ hprev,
    float* __restrict__ hout, __bf16* __restrict__ houtbf,
    float* __restrict__ hout2,
    int bt, int wave, int lane)
{
  const int l15 = lane & 15;
  const int kh  = lane >> 4;
  const int j0a = (wave * 2) * 16;
  const int j0b = j0a + 16;

  const float*  arf = (const float*)a0v + (A0F32 ? (size_t)(bt * 16 + l15) * lda0 : 0);
  const __bf16* arb = (const __bf16*)a0v + (A0F32 ? 0 : (size_t)(bt * 16 + l15) * lda0);
  const __bf16* ar1 = a1bf + (size_t)(bt * 16 + l15) * lda1;

  v8f acr0 = {}, acr1 = {}, acz0 = {}, acz1 = {}, acn0 = {}, acn1 = {};

  for (int kk = 0; kk < ktot; kk += 32) {
    // region choice is uniform per chunk: k0len and kk are multiples of 32
    v16bf afrag;
    if (kk < k0len) {
      if (A0F32) afrag = frag_from_f32(arf + kk + kh * 16, arf + kk + kh * 16 + 8);
      else       afrag = frag_from_bf16(arb + kk + kh * 16, arb + kk + kh * 16 + 8);
    } else {
      const int kr = kk - k0len;
      afrag = frag_from_bf16(ar1 + kr + kh * 16, ar1 + kr + kh * 16 + 8);
    }

    // B fragment (32x16): lane l holds col n=l&15, K = kk + (l>>4)*16 + e
    const int koff = kk + kh * 16;
    const v16bf br0 = *(const v16bf*)(wrz + (size_t)(j0a + l15) * ldrz + koff);
    const v16bf br1 = *(const v16bf*)(wrz + (size_t)(j0b + l15) * ldrz + koff);
    const v16bf bz0 = *(const v16bf*)(wrz + (size_t)(HH + j0a + l15) * ldrz + koff);
    const v16bf bz1 = *(const v16bf*)(wrz + (size_t)(HH + j0b + l15) * ldrz + koff);
    acr0 = wmma_bf16(afrag, br0, acr0);
    acr1 = wmma_bf16(afrag, br1, acr1);
    acz0 = wmma_bf16(afrag, bz0, acz0);
    acz1 = wmma_bf16(afrag, bz1, acz1);
    if (kk < k0len) {              // n-GEMM covers only the a0 part of A
      const v16bf bn0 = *(const v16bf*)(wn + (size_t)(j0a + l15) * ldn + koff);
      const v16bf bn1 = *(const v16bf*)(wn + (size_t)(j0b + l15) * ldn + koff);
      acn0 = wmma_bf16(afrag, bn0, acn0);
      acn1 = wmma_bf16(afrag, bn1, acn1);
    }
  }

  // C/D layout: vgpr v <-> row m = v + 8*(lane>>4), col n = lane&15
  auto epi = [&](v8f vr, v8f vz, v8f vn, int j0) {
    const int col = j0 + l15;
#pragma unroll
    for (int v = 0; v < 8; ++v) {
      const int row = bt * 16 + kh * 8 + v;
      const float hp = hprev[(size_t)row * HH + col];
      const float rg = fsigmoid(vr[v]);
      const float zg = fsigmoid(vz[v]);
      const float ng = ftanh_fast(vn[v] + rg * hp);
      const float hn = (1.0f - zg) * ng + zg * hp;
      hout[(size_t)row * HH + col]   = hn;
      houtbf[(size_t)row * HH + col] = f2bf(hn);
      if (hout2) hout2[(size_t)row * HH + col] = hn;
    }
  };
  epi(acr0, acz0, acn0, j0a);
  epi(acr1, acz1, acn1, j0b);
}

__device__ __forceinline__ void grid_barrier(unsigned* cnt, unsigned round) {
  __syncthreads();
  if (threadIdx.x == 0) {
    __threadfence();   // release our stores to device scope
    __hip_atomic_fetch_add(cnt, 1u, __ATOMIC_RELEASE, __HIP_MEMORY_SCOPE_AGENT);
    const unsigned target = (round + 1u) * (unsigned)NBLK;
    while (__hip_atomic_load(cnt, __ATOMIC_ACQUIRE, __HIP_MEMORY_SCOPE_AGENT)
           < target) {
      __builtin_amdgcn_s_sleep(2);
    }
  }
  __syncthreads();
  __threadfence();     // acquire other blocks' stores
}

// Weight repack fp32->bf16 (+ gate concat) and state seeding.
__global__ void gru_prep(const float* __restrict__ Wi0, const float* __restrict__ Wh0,
                         const float* __restrict__ Wn0, const float* __restrict__ Wi1,
                         const float* __restrict__ Wh1, const float* __restrict__ Wn1,
                         const float* __restrict__ state,
                         __bf16* __restrict__ wrz0, __bf16* __restrict__ wn0,
                         __bf16* __restrict__ wrz1, __bf16* __restrict__ wn1,
                         float* __restrict__ h0buf,
                         __bf16* __restrict__ h0bf, __bf16* __restrict__ h1bf)
{
  const int total = N_RZ0 + N_N0 + N_RZ1 + N_N1 + 2 * N_H + 2 * N_H;
  for (int id = blockIdx.x * blockDim.x + threadIdx.x; id < total;
       id += gridDim.x * blockDim.x) {
    int i = id;
    if (i < N_RZ0) {                       // Wrz0[j][k] = k<I ? Wi0[j][k] : Wh0[j][k-I]
      const int j = i / (II + HH), k = i % (II + HH);
      wrz0[i] = f2bf(k < II ? Wi0[(size_t)j * II + k]
                            : Wh0[(size_t)j * HH + (k - II)]);
    } else if ((i -= N_RZ0) < N_N0) {      // Wn0 straight cast
      wn0[i] = f2bf(Wn0[i]);
    } else if ((i -= N_N0) < N_RZ1) {      // Wrz1[j][k] = k<H ? Wi1[j][k] : Wh1[j][k-H]
      const int j = i / (2 * HH), k = i % (2 * HH);
      wrz1[i] = f2bf(k < HH ? Wi1[(size_t)j * HH + k]
                            : Wh1[(size_t)j * HH + (k - HH)]);
    } else if ((i -= N_RZ1) < N_N1) {      // Wn1 straight cast
      wn1[i] = f2bf(Wn1[i]);
    } else if ((i -= N_N1) < 2 * N_H) {    // h0 fp32 ping-pong: [0]=0, [1]=state[0]
      h0buf[i] = (i < N_H) ? 0.0f : state[i - N_H];
    } else {                               // bf16 mirrors: h0bf[1]=state0, h1bf[0]=state1
      i -= 2 * N_H;
      if (i < N_H) h0bf[N_H + i] = f2bf(state[i]);
      else         h1bf[i - N_H] = f2bf(state[N_H + (i - N_H)]);
    }
  }
}

__global__ void __launch_bounds__(256, 1)
gru_persistent(const float* __restrict__ x, const float* __restrict__ state,
               const __bf16* __restrict__ wrz0, const __bf16* __restrict__ wn0,
               const __bf16* __restrict__ wrz1, const __bf16* __restrict__ wn1,
               float* __restrict__ h0buf, __bf16* __restrict__ h0bf,
               __bf16* __restrict__ h1bf, float* __restrict__ out,
               unsigned* __restrict__ barrier_cnt)
{
  const int layer = blockIdx.x >> 4;       // 0: layer0, 1: layer1
  const int bt    = blockIdx.x & 15;       // batch tile (16 rows)
  const int wave  = threadIdx.x >> 5;
  const int lane  = threadIdx.x & 31;

  for (int r = 0; r <= T_STEPS; ++r) {
    if (layer == 0) {
      if (r < T_STEPS) {
        // h0^r = f0(x_r, h0^{r-1}); parity: write r&1, read (r+1)&1
        const size_t pw = (size_t)(r & 1) * N_H, pr = (size_t)((r + 1) & 1) * N_H;
        const float* xt = x + (size_t)r * BB * II;
        float* fin = (r == T_STEPS - 1) ? (out + (size_t)T_STEPS * N_H) : nullptr;
        gru_cell_tile<true>(xt, II, II, h0bf + pr, HH, II + HH,
                            wrz0, II + HH, wn0, II,
                            h0buf + pr, h0buf + pw, h0bf + pw, fin,
                            bt, wave, lane);
      }
    } else {
      if (r >= 1) {
        // h1^{r-1} = f1(h0^{r-1}, h1^{r-2}); fp32 h1 history lives in `out`
        const int t = r - 1;
        const size_t pw = (size_t)(r & 1) * N_H, pr = (size_t)((r + 1) & 1) * N_H;
        const float* h1prevf = (r == 1) ? (state + (size_t)N_H)
                                        : (out + (size_t)(r - 2) * N_H);
        float* fin = (r == T_STEPS) ? (out + (size_t)T_STEPS * N_H + N_H) : nullptr;
        gru_cell_tile<false>(h0bf + pr, HH, HH, h1bf + pr, HH, HH + HH,
                             wrz1, HH + HH, wn1, HH,
                             h1prevf, out + (size_t)t * N_H, h1bf + pw, fin,
                             bt, wave, lane);
      }
    }
    grid_barrier(barrier_cnt, (unsigned)r);
  }
}

extern "C" void kernel_launch(void* const* d_in, const int* in_sizes, int n_in,
                              void* d_out, int out_size, void* d_ws, size_t ws_size,
                              hipStream_t stream) {
  (void)in_sizes; (void)n_in; (void)out_size; (void)ws_size;
  const float* x     = (const float*)d_in[0];
  const float* state = (const float*)d_in[1];
  const float* Wi0   = (const float*)d_in[2];
  const float* Wh0   = (const float*)d_in[3];
  const float* Wn0   = (const float*)d_in[4];
  const float* Wi1   = (const float*)d_in[5];
  const float* Wh1   = (const float*)d_in[6];
  const float* Wn1   = (const float*)d_in[7];
  float* out = (float*)d_out;

  // ws layout (256B aligned): [cnt][wrz0][wn0][wrz1][wn1][h0 f32 x2][h0bf x2][h1bf x2]
  char* ws = (char*)d_ws;
  unsigned* cnt = (unsigned*)ws;
  __bf16* wrz0 = (__bf16*)(ws + 256);
  __bf16* wn0  = wrz0 + N_RZ0;
  __bf16* wrz1 = wn0  + N_N0;
  __bf16* wn1  = wrz1 + N_RZ1;
  float*  h0buf = (float*)(ws + 256 + 2 * (size_t)(N_RZ0 + N_N0 + N_RZ1 + N_N1));
  __bf16* h0bf  = (__bf16*)(h0buf + 2 * (size_t)N_H);
  __bf16* h1bf  = h0bf + 2 * (size_t)N_H;

  (void)hipMemsetAsync(d_ws, 0, 256, stream);  // zero barrier counter each call
  gru_prep<<<1600, 256, 0, stream>>>(Wi0, Wh0, Wn0, Wi1, Wh1, Wn1, state,
                                     wrz0, wn0, wrz1, wn1, h0buf, h0bf, h1bf);
  gru_persistent<<<NBLK, 256, 0, stream>>>(x, state, wrz0, wn0, wrz1, wn1,
                                           h0buf, h0bf, h1bf, out, cnt);
}
